// residual_49108656062731
// MI455X (gfx1250) — matrix operand order repaired and checked by reading, not compiled, not run on hardware
//
#include <hip/hip_runtime.h>

typedef __attribute__((ext_vector_type(16))) _Float16 v16h;
typedef __attribute__((ext_vector_type(8)))  _Float16 v8h;
typedef __attribute__((ext_vector_type(8)))  float    v8f;
typedef __attribute__((ext_vector_type(4)))  unsigned int u32x4;
typedef __attribute__((ext_vector_type(8)))  int          i32x8;
typedef __attribute__((ext_vector_type(4)))  int          i32x4;

#define BATCH 8
#define SEQ   4096
#define DIM   1024
#define HID   128
#define QKVW  384
#define ROWS  (BATCH * SEQ)   // 32768

#ifndef __has_builtin
#define __has_builtin(x) 0
#endif
#if __has_builtin(__builtin_amdgcn_tensor_load_to_lds) && \
    __has_builtin(__builtin_amdgcn_s_wait_tensorcnt)
#define USE_TDM 1
#else
#define USE_TDM 0
#endif

union HV16 { v16h v; v8h h[2]; };

// A fragment (16x32 f16): lane = {sel,m}; halves e -> K = kb + sel*16 + e
// (contiguous order, matches load_b_frag)
__device__ __forceinline__ v16h load_a_frag(const _Float16* base, int row_stride,
                                            int m, int kb, int sel) {
  const _Float16* p = base + m * row_stride + kb + sel * 16;
  HV16 u;
  u.h[0] = *(const v8h*)(p);
  u.h[1] = *(const v8h*)(p + 8);
  return u.v;
}

// B fragment (32x16 f16) from an [N][K] (column-major) buffer: lane n = lane&15,
// halves e -> K = kb + sel*16 + e  => 16 contiguous halves (32B) per lane.
__device__ __forceinline__ v16h load_b_frag(const _Float16* colbase, int kb, int sel) {
  return *(const v16h*)(colbase + kb + sel * 16);
}

// ---------------- kernel 1: cast + transpose weights to f16 ----------------
__global__ void prep_weights_kernel(const float* __restrict__ wqkv,
                                    const float* __restrict__ wout,
                                    _Float16* __restrict__ wqkv_t,   // [384][1024]
                                    _Float16* __restrict__ wout_t) { // [1024][128]
  int i = blockIdx.x * blockDim.x + threadIdx.x;
  if (i < DIM * QKVW) {          // wqkv: [1024][384]
    int c = i / QKVW, k = i % QKVW;
    wqkv_t[k * DIM + c] = (_Float16)wqkv[i];
  }
  if (i < HID * DIM) {           // wout: [128][1024]
    int k = i / DIM, n = i % DIM;
    wout_t[n * HID + k] = (_Float16)wout[i];
  }
}

// ---------------- kernel 2: LayerNorm + cast to f16 ----------------
__global__ void ln_cast_kernel(const float* __restrict__ x,
                               const float* __restrict__ gamma,
                               const float* __restrict__ beta,
                               _Float16* __restrict__ xn16) {
  __shared__ float red[24];
  int r = blockIdx.x;
  const float* xr = x + (size_t)r * DIM;
  float vals[4];
  float s = 0.f, s2 = 0.f;
  for (int j = 0; j < 4; ++j) {
    float v = xr[threadIdx.x + j * 256];
    vals[j] = v; s += v; s2 += v * v;
  }
  for (int off = 16; off > 0; off >>= 1) {
    s  += __shfl_down(s,  off, 32);
    s2 += __shfl_down(s2, off, 32);
  }
  int wave = threadIdx.x >> 5, lane = threadIdx.x & 31;
  if (lane == 0) { red[wave] = s; red[8 + wave] = s2; }
  __syncthreads();
  if (threadIdx.x == 0) {
    float ts = 0.f, ts2 = 0.f;
    for (int w = 0; w < 8; ++w) { ts += red[w]; ts2 += red[8 + w]; }
    float mean = ts * (1.0f / DIM);
    float var  = ts2 * (1.0f / DIM) - mean * mean;
    red[16] = mean;
    red[17] = rsqrtf(var + 1e-5f);
  }
  __syncthreads();
  float mean = red[16], rstd = red[17];
  for (int j = 0; j < 4; ++j) {
    int c = threadIdx.x + j * 256;
    float y = (vals[j] - mean) * rstd * gamma[c] + beta[c];
    xn16[(size_t)r * DIM + c] = (_Float16)y;
  }
}

// ---------------- kernel 3: QKV GEMM (M=32768,K=1024,N=384), WMMA + TDM ----------------
__global__ void qkv_gemm_kernel(const _Float16* __restrict__ xn16,
                                const _Float16* __restrict__ wqkv_t,
                                float* __restrict__ qkv) {
  __shared__ _Float16 As[16][DIM];   // 32 KB, LDS offset 0 (only shared array)
  int r0 = blockIdx.x * 16;

#if USE_TDM
  // Fill the 16x1024-half (32 KB, fully contiguous) A panel with one TDM
  // descriptor: 1-row tile of 4096 x 8-byte elements. D# per ISA 8.3/8.4.
  if (threadIdx.x < 32) {            // one wave issues; EXEC ignored by TDM
    unsigned long long gaddr =
        (unsigned long long)(size_t)(xn16 + (size_t)r0 * DIM);
    u32x4 g0;
    g0[0] = 1u;                                   // count=1 (valid), user mode
    g0[1] = 0u;                                   // lds_addr = 0
    g0[2] = (unsigned)(gaddr & 0xffffffffu);      // global_addr[31:0]
    g0[3] = ((unsigned)(gaddr >> 32) & 0x01ffffffu) | (2u << 30); // [56:32]|type=2
    const unsigned td0 = 4096u;                   // 4096 x 8B = 32 KB
    i32x8 g1;
    g1[0] = (int)(3u << 16);                      // wg_mask=0, data_size=3 (8B)
    g1[1] = (int)((td0 & 0xffffu) << 16);         // tensor_dim0[15:0]
    g1[2] = (int)(((td0 >> 16) & 0xffffu) | (1u << 16)); // td0 hi | tensor_dim1=1
    g1[3] = (int)((td0 & 0xffffu) << 16);         // tile_dim0 = 4096
    g1[4] = 1;                                    // tile_dim1=1, tile_dim2=0
    g1[5] = (int)td0;                             // tensor_dim0_stride lo32
    g1[6] = 0;
    g1[7] = 0;
    i32x4 g2 = {}; i32x4 g3 = {};
#if __clang_major__ >= 23
    i32x8 g1e = {};
    __builtin_amdgcn_tensor_load_to_lds(g0, g1, g2, g3, g1e, 0);
#else
    __builtin_amdgcn_tensor_load_to_lds(g0, g1, g2, g3, 0);
#endif
    __builtin_amdgcn_s_wait_tensorcnt(0);
  }
  // The TDM descriptor carries the LDS address only as an integer, so LLVM
  // cannot see that `As` is written. Escape As's address into a memory-
  // clobbering asm so the asm is a potential writer of As; otherwise the
  // ds_loads of As fold to undef and get deleted (observed in round 2/3).
  asm volatile("" :: "v"(&As[0][0]) : "memory");
#else
  {
    const v8h* src = (const v8h*)(xn16 + (size_t)r0 * DIM);
    v8h* dst = (v8h*)(&As[0][0]);
    for (int j = 0; j < 8; ++j) {
      __builtin_prefetch(src + threadIdx.x + j * 256 + 2048, 0, 1);
      dst[threadIdx.x + j * 256] = src[threadIdx.x + j * 256];
    }
  }
#endif
  __syncthreads();

  int wave = threadIdx.x >> 5, lane = threadIdx.x & 31;
  int m = lane & 15, sel = lane >> 4;
  for (int t = 0; t < 3; ++t) {
    int nt = wave * 3 + t;                       // 24 N-tiles of 16
    int ncol = nt * 16 + m;
    const _Float16* bcol = wqkv_t + (size_t)ncol * DIM;
    v8f acc = {};
    for (int kc = 0; kc < DIM; kc += 32) {
      v16h a = load_a_frag(&As[0][0], DIM, m, kc, sel);
      v16h b = load_b_frag(bcol, kc, sel);
      acc = __builtin_amdgcn_wmma_f32_16x16x32_f16(false, a, false, b,
                                                   (short)0, acc, false, false);
    }
    for (int v = 0; v < 8; ++v) {
      int row = r0 + v + sel * 8;
      qkv[(size_t)row * QKVW + ncol] = acc[v];
    }
  }
}

// ---------------- kernel 4: softmax stats over sequence for k ----------------
__global__ void ksoftmax_stats_kernel(const float* __restrict__ qkv,
                                      float* __restrict__ kmax,
                                      float* __restrict__ ksum) {
  __shared__ float red[8];
  int b = blockIdx.x >> 7;          // 0..7
  int c = blockIdx.x & 127;         // k channel = h*32 + d
  const float* base = qkv + (size_t)b * SEQ * QKVW + HID + c;
  int wave = threadIdx.x >> 5, lane = threadIdx.x & 31;

  float mx = -1e30f;
  for (int n = threadIdx.x; n < SEQ; n += 256)
    mx = fmaxf(mx, base[(size_t)n * QKVW]);
  for (int off = 16; off > 0; off >>= 1)
    mx = fmaxf(mx, __shfl_down(mx, off, 32));
  if (lane == 0) red[wave] = mx;
  __syncthreads();
  float m8 = red[0];
  for (int w = 1; w < 8; ++w) m8 = fmaxf(m8, red[w]);
  __syncthreads();

  float sm = 0.f;
  for (int n = threadIdx.x; n < SEQ; n += 256)
    sm += __expf(base[(size_t)n * QKVW] - m8);
  for (int off = 16; off > 0; off >>= 1)
    sm += __shfl_down(sm, off, 32);
  if (lane == 0) red[wave] = sm;
  __syncthreads();
  if (threadIdx.x == 0) {
    float t = 0.f;
    for (int w = 0; w < 8; ++w) t += red[w];
    kmax[blockIdx.x] = m8;
    ksum[blockIdx.x] = t;
  }
}

// ---------------- kernel 5: context[b,h] = softmax(K)^T V (32x32), WMMA ----------------
__global__ void context_kernel(const float* __restrict__ qkv,
                               const float* __restrict__ kmax,
                               const float* __restrict__ ksum,
                               _Float16* __restrict__ ctx_t) { // [b][h][e][d] f16
  __shared__ _Float16 Ks[32][32];   // [d][n]
  __shared__ _Float16 Vs[32][32];   // [e][n]
  int b = blockIdx.x >> 2, h = blockIdx.x & 3;
  int wave = threadIdx.x >> 5, lane = threadIdx.x & 31;
  int dq = wave >> 1, eq = wave & 1;          // 16x16 quadrant per wave
  int mL = lane & 15, sel = lane >> 4;

  v8f acc = {};
  for (int nb = 0; nb < SEQ; nb += 32) {
    __syncthreads();
    for (int i = threadIdx.x; i < 1024; i += 128) {
      int d = i & 31, n = i >> 5;
      size_t rowoff = ((size_t)b * SEQ + nb + n) * QKVW;
      float kv = qkv[rowoff + HID + h * 32 + d];
      int sc = b * 128 + h * 32 + d;
      float kn = __expf(kv - kmax[sc]) / ksum[sc];
      Ks[d][n] = (_Float16)kn;
      Vs[d][n] = (_Float16)qkv[rowoff + 2 * HID + h * 32 + d]; // e-index reuses d slot
    }
    __syncthreads();
    v16h a = load_a_frag(&Ks[dq * 16][0], 32, mL, 0, sel);
    v16h bfrag = load_b_frag(&Vs[eq * 16 + mL][0], 0, sel);
    acc = __builtin_amdgcn_wmma_f32_16x16x32_f16(false, a, false, bfrag,
                                                 (short)0, acc, false, false);
  }
  int e_g = eq * 16 + mL;
  for (int v = 0; v < 8; ++v) {
    int d_g = dq * 16 + v + sel * 8;
    ctx_t[(((size_t)(b * 4 + h) * 32) + e_g) * 32 + d_g] = (_Float16)acc[v];
  }
}

// ---------------- kernel 6: out = (q·ctx)·w_out^T + b_out + x, WMMA ----------------
__global__ void out_kernel(const float* __restrict__ qkv,
                           const _Float16* __restrict__ ctx_t,
                           const _Float16* __restrict__ wout_t,
                           const float* __restrict__ b_out,
                           const float* __restrict__ x,
                           float* __restrict__ out) {
  __shared__ _Float16 Qs[16][HID];
  __shared__ _Float16 Us[16][HID];
  int r0 = blockIdx.x * 16;
  int b = r0 >> 12;                    // 16 | 4096 so whole block is same batch
  const float qscale = 0.17677669529663689f;  // 32^-0.5
  for (int i = threadIdx.x; i < 16 * HID; i += 256) {
    int row = i >> 7, c = i & 127;
    Qs[row][c] = (_Float16)(qkv[(size_t)(r0 + row) * QKVW + c] * qscale);
  }
  __syncthreads();
  int wave = threadIdx.x >> 5, lane = threadIdx.x & 31;
  int mL = lane & 15, sel = lane >> 4;
  {
    // u = q_h (16x32) @ ctx_h (32x32): one WMMA per (head, 16-col half) = 8 waves
    int h = wave >> 1, jn = wave & 1;
    v16h a = load_a_frag(&Qs[0][0], HID, mL, h * 32, sel);
    const _Float16* bp = ctx_t +
        (size_t)((b * 4 + h) * 32 + jn * 16 + mL) * 32;    // [e][d], K=d contiguous
    v16h bfrag = load_b_frag(bp, 0, sel);
    v8f u = {};
    u = __builtin_amdgcn_wmma_f32_16x16x32_f16(false, a, false, bfrag,
                                               (short)0, u, false, false);
    for (int v = 0; v < 8; ++v)
      Us[v + sel * 8][h * 32 + jn * 16 + mL] = (_Float16)u[v];
  }
  __syncthreads();
  // y = u (16x128) @ w_out (128x1024) ; each wave 8 N-tiles
  for (int t = 0; t < 8; ++t) {
    int nt = wave * 8 + t;
    int ncol = nt * 16 + mL;
    const _Float16* bcol = wout_t + (size_t)ncol * HID;
    v8f acc = {};
    for (int kc = 0; kc < HID; kc += 32) {
      v16h a = load_a_frag(&Us[0][0], HID, mL, kc, sel);
      v16h bfrag = load_b_frag(bcol, kc, sel);
      acc = __builtin_amdgcn_wmma_f32_16x16x32_f16(false, a, false, bfrag,
                                                   (short)0, acc, false, false);
    }
    float bo = b_out[ncol];
    for (int v = 0; v < 8; ++v) {
      size_t row = (size_t)r0 + v + sel * 8;
      out[row * DIM + ncol] = acc[v] + bo + x[row * DIM + ncol];
    }
  }
}

extern "C" void kernel_launch(void* const* d_in, const int* in_sizes, int n_in,
                              void* d_out, int out_size, void* d_ws, size_t ws_size,
                              hipStream_t stream) {
  (void)in_sizes; (void)n_in; (void)out_size; (void)ws_size;
  const float* x     = (const float*)d_in[0];
  const float* gamma = (const float*)d_in[1];
  const float* beta  = (const float*)d_in[2];
  const float* wqkv  = (const float*)d_in[3];
  const float* wout  = (const float*)d_in[4];
  const float* bout  = (const float*)d_in[5];
  float* out = (float*)d_out;

  char* ws = (char*)d_ws;
  size_t off = 0;
  auto alloc = [&](size_t bytes) {
    void* p = ws + off;
    off += (bytes + 255) & ~(size_t)255;
    return p;
  };
  _Float16* xn16   = (_Float16*)alloc((size_t)ROWS * DIM * 2);   // 64 MB
  _Float16* wqkv_t = (_Float16*)alloc((size_t)QKVW * DIM * 2);
  _Float16* wout_t = (_Float16*)alloc((size_t)DIM * HID * 2);
  float*    qkv    = (float*)   alloc((size_t)ROWS * QKVW * 4);  // 48 MB
  float*    kmax   = (float*)   alloc(BATCH * HID * 4);
  float*    ksum   = (float*)   alloc(BATCH * HID * 4);
  _Float16* ctx_t  = (_Float16*)alloc((size_t)BATCH * 4 * 32 * 32 * 2);

  prep_weights_kernel<<<dim3((DIM * QKVW + 255) / 256), dim3(256), 0, stream>>>(
      wqkv, wout, wqkv_t, wout_t);
  ln_cast_kernel<<<dim3(ROWS), dim3(256), 0, stream>>>(x, gamma, beta, xn16);
  qkv_gemm_kernel<<<dim3(ROWS / 16), dim3(256), 0, stream>>>(xn16, wqkv_t, qkv);
  ksoftmax_stats_kernel<<<dim3(BATCH * HID), dim3(256), 0, stream>>>(qkv, kmax, ksum);
  context_kernel<<<dim3(BATCH * 4), dim3(128), 0, stream>>>(qkv, kmax, ksum, ctx_t);
  out_kernel<<<dim3(ROWS / 16), dim3(256), 0, stream>>>(qkv, ctx_t, wout_t, bout, x, out);
}
